// ConditionalGAT_11553462026721
// MI455X (gfx1250) — matrix operand order repaired and checked by reading, not compiled, not run on hardware
//
#include <hip/hip_runtime.h>
#include <hip/hip_bf16.h>
#include <stdint.h>

#define IN_CH 768
#define CONDC 768
#define HIDC  128
#define HEADS 4
#define F1 (HEADS*HIDC)   /* 512 */

typedef __attribute__((ext_vector_type(16))) __bf16        bf16x16;
typedef __attribute__((ext_vector_type(8)))  float         f32x8;
typedef __attribute__((ext_vector_type(8)))  unsigned short u16x8;

union FragU { u16x8 h[2]; bf16x16 v; };

__device__ __forceinline__ unsigned short f2bf(float f) {
  unsigned u = __float_as_uint(f);
  u += 0x7FFFu + ((u >> 16) & 1u);          // round-to-nearest-even
  return (unsigned short)(u >> 16);
}
__device__ __forceinline__ unsigned fenc(float f) {       // monotonic float->u32
  unsigned u = __float_as_uint(f);
  return (u & 0x80000000u) ? ~u : (u | 0x80000000u);
}
__device__ __forceinline__ float fdec(unsigned u) {
  unsigned b = (u & 0x80000000u) ? (u & 0x7FFFFFFFu) : ~u;
  return __uint_as_float(b);
}

// ---------------- generic fill ----------------
__global__ void k_fill_u32(unsigned* __restrict__ p, unsigned val, long long n) {
  long long t = (long long)blockIdx.x * blockDim.x + threadIdx.x;
  if (t < n) p[t] = val;
}

// ---------------- weight convert + transpose: W[K][Nout] f32 -> Wt[Nout][K] bf16 ----------------
__global__ void k_convert_transpose(const float* __restrict__ W, unsigned short* __restrict__ Wt,
                                    int K, int Nout) {
  long long t = (long long)blockIdx.x * blockDim.x + threadIdx.x;
  if (t >= (long long)K * Nout) return;
  int n = (int)(t % Nout), k = (int)(t / Nout);
  Wt[(size_t)n * K + k] = f2bf(W[t]);
}

// ---------------- A1 = [x | cond[batch]] in bf16, [N][1536] ----------------
__global__ void k_build_A1(const float* __restrict__ x, const float* __restrict__ cond,
                           const int* __restrict__ batch, unsigned short* __restrict__ A1, int N) {
  long long t = (long long)blockIdx.x * blockDim.x + threadIdx.x;
  long long total = (long long)N * (IN_CH + CONDC);
  if (t >= total) return;
  int c = (int)(t % (IN_CH + CONDC));
  int n = (int)(t / (IN_CH + CONDC));
  float v = (c < IN_CH) ? x[(size_t)n * IN_CH + c]
                        : cond[(size_t)batch[n] * CONDC + (c - IN_CH)];
  A1[t] = f2bf(v);
}

// ---------------- WMMA bf16 GEMM: C[M][Nout] = A[M][K] * Bt[Nout][K]^T ----------------
// block = 256 threads = 8 waves (4 along M x 2 along N); wave tile 32x32 (2x2 wmma)
__global__ __launch_bounds__(256) void k_gemm_wmma(
    const unsigned short* __restrict__ A, const unsigned short* __restrict__ Bt,
    float* __restrict__ C, int M, int K, int Nout) {
  const int lane = threadIdx.x & 31;
  const int wave = threadIdx.x >> 5;
  const int wm = wave & 3, wn = wave >> 2;
  const int m0 = blockIdx.x * 128 + wm * 32;
  const int n0 = blockIdx.y * 64  + wn * 32;

  f32x8 acc[2][2] = {};
  const int kbA = (lane < 16) ? 0 : 8;    // A frag: K = kbA..kbA+7, 16+kbA..16+kbA+7
  const int kbB = (lane < 16) ? 0 : 8;    // B frag: K = kbB..kbB+7, 16+kbB..16+kbB+7
  const int rA0 = min(m0 + (lane & 15), M - 1);
  const int rA1 = min(m0 + 16 + (lane & 15), M - 1);
  const unsigned short* pA0 = A + (size_t)rA0 * K;
  const unsigned short* pA1 = A + (size_t)rA1 * K;
  const unsigned short* pB0 = Bt + (size_t)(n0 + (lane & 15)) * K;
  const unsigned short* pB1 = Bt + (size_t)(n0 + 16 + (lane & 15)) * K;

  for (int k = 0; k < K; k += 32) {
    FragU a0, a1, b0, b1;
    a0.h[0] = *(const u16x8*)(pA0 + k + kbA);
    a0.h[1] = *(const u16x8*)(pA0 + k + 16 + kbA);
    a1.h[0] = *(const u16x8*)(pA1 + k + kbA);
    a1.h[1] = *(const u16x8*)(pA1 + k + 16 + kbA);
    b0.h[0] = *(const u16x8*)(pB0 + k + kbB);
    b0.h[1] = *(const u16x8*)(pB0 + k + 16 + kbB);
    b1.h[0] = *(const u16x8*)(pB1 + k + kbB);
    b1.h[1] = *(const u16x8*)(pB1 + k + 16 + kbB);
    acc[0][0] = __builtin_amdgcn_wmma_f32_16x16x32_bf16(false, a0.v, false, b0.v, (short)0, acc[0][0], false, false);
    acc[0][1] = __builtin_amdgcn_wmma_f32_16x16x32_bf16(false, a0.v, false, b1.v, (short)0, acc[0][1], false, false);
    acc[1][0] = __builtin_amdgcn_wmma_f32_16x16x32_bf16(false, a1.v, false, b0.v, (short)0, acc[1][0], false, false);
    acc[1][1] = __builtin_amdgcn_wmma_f32_16x16x32_bf16(false, a1.v, false, b1.v, (short)0, acc[1][1], false, false);
  }

  const int rowHalf = (lane >> 4) * 8;
  const int col = lane & 15;
  for (int i = 0; i < 2; ++i)
    for (int j = 0; j < 2; ++j)
      for (int v = 0; v < 8; ++v) {
        int r = m0 + 16 * i + rowHalf + v;
        if (r < M) C[(size_t)r * Nout + n0 + 16 * j + col] = acc[i][j][v];
      }
}

// ---------------- per-node attention logits: a_s/a_d [N*H] ----------------
__global__ void k_node_logits(const float* __restrict__ Hpre, const float* __restrict__ att_s,
                              const float* __restrict__ att_d, float* __restrict__ a_s,
                              float* __restrict__ a_d, int N, int H, int D) {
  int t = blockIdx.x * blockDim.x + threadIdx.x;
  if (t >= N * H) return;
  int h = t % H, n = t / H;
  const float* hp = Hpre + ((size_t)n * H + h) * D;
  float ss = 0.f, dd = 0.f;
  for (int d = 0; d < D; ++d) { float v = hp[d]; ss += v * att_s[h * D + d]; dd += v * att_d[h * D + d]; }
  a_s[t] = ss; a_d[t] = dd;
}

// ---------------- edge pass 1: segment_max via encoded atomicMax ----------------
__global__ void k_edge_max(const int* __restrict__ ei, int E, int N,
                           const float* __restrict__ a_s, const float* __restrict__ a_d,
                           unsigned* __restrict__ menc, int H) {
  long long t = (long long)blockIdx.x * blockDim.x + threadIdx.x;
  long long tot = (long long)(E + N) * H;
  if (t >= tot) return;
  int h = (int)(t % H); int e = (int)(t / H);
  int s = (e < E) ? ei[e] : (e - E);
  int d = (e < E) ? ei[E + e] : (e - E);
  float l = a_s[(size_t)s * H + h] + a_d[(size_t)d * H + h];
  l = (l > 0.f) ? l : 0.2f * l;                       // leaky_relu 0.2
  atomicMax(&menc[(size_t)d * H + h], fenc(l));
}

// ---------------- edge pass 2: p=exp(l-max), segment_sum ----------------
__global__ void k_edge_expsum(const int* __restrict__ ei, int E, int N,
                              const float* __restrict__ a_s, const float* __restrict__ a_d,
                              const unsigned* __restrict__ menc, float* __restrict__ pbuf,
                              float* __restrict__ ssum, int H) {
  long long t = (long long)blockIdx.x * blockDim.x + threadIdx.x;
  long long tot = (long long)(E + N) * H;
  if (t >= tot) return;
  int h = (int)(t % H); int e = (int)(t / H);
  int s = (e < E) ? ei[e] : (e - E);
  int d = (e < E) ? ei[E + e] : (e - E);
  float l = a_s[(size_t)s * H + h] + a_d[(size_t)d * H + h];
  l = (l > 0.f) ? l : 0.2f * l;
  float p = __expf(l - fdec(menc[(size_t)d * H + h]));
  pbuf[t] = p;
  atomicAdd(&ssum[(size_t)d * H + h], p);
}

// ---------------- edge pass 3: weighted scatter (float4 chunks, atomics land in L2) ----------------
__global__ void k_edge_agg(const int* __restrict__ ei, int E, int N,
                           const float* __restrict__ Hpre, const float* __restrict__ pbuf,
                           const float* __restrict__ ssum, float* __restrict__ agg, int H, int D) {
  long long t = (long long)blockIdx.x * blockDim.x + threadIdx.x;
  int chunks = D >> 2;
  long long tot = (long long)(E + N) * H * chunks;
  if (t >= tot) return;
  int c = (int)(t % chunks);
  int h = (int)((t / chunks) % H);
  int e = (int)(t / ((long long)chunks * H));
  int s = (e < E) ? ei[e] : (e - E);
  int d = (e < E) ? ei[E + e] : (e - E);
  float w = pbuf[(size_t)e * H + h] / (ssum[(size_t)d * H + h] + 1e-16f);
  const float4 hv = *(const float4*)(Hpre + ((size_t)s * H + h) * D + 4 * c);
  float* ap = agg + ((size_t)d * H + h) * D + 4 * c;
  atomicAdd(ap + 0, hv.x * w);
  atomicAdd(ap + 1, hv.y * w);
  atomicAdd(ap + 2, hv.z * w);
  atomicAdd(ap + 3, hv.w * w);
}

// ---------------- bias + relu -> bf16 (feeds next GEMM) ----------------
__global__ void k_bias_relu_bf16(const float* __restrict__ agg, const float* __restrict__ bias,
                                 unsigned short* __restrict__ Hbf, int N, int F) {
  long long t = (long long)blockIdx.x * blockDim.x + threadIdx.x;
  if (t >= (long long)N * F) return;
  int f = (int)(t % F);
  float v = agg[t] + bias[f];
  Hbf[t] = f2bf(v > 0.f ? v : 0.f);
}

// ---------------- head: out[n] = relu(agg2[n]+b2) . out_W + out_b ----------------
__global__ void k_final(const float* __restrict__ agg2, const float* __restrict__ b2,
                        const float* __restrict__ out_W, const float* __restrict__ out_b,
                        float* __restrict__ out, int N) {
  int n = blockIdx.x * blockDim.x + threadIdx.x;
  if (n >= N) return;
  float acc = out_b[0];
  const float* hp = agg2 + (size_t)n * HIDC;
  for (int f = 0; f < HIDC; ++f) {
    float v = hp[f] + b2[f];
    v = v > 0.f ? v : 0.f;
    acc += v * out_W[f];
  }
  out[n] = acc;
}

extern "C" void kernel_launch(void* const* d_in, const int* in_sizes, int n_in,
                              void* d_out, int out_size, void* d_ws, size_t ws_size,
                              hipStream_t stream) {
  const float* x      = (const float*)d_in[0];
  const float* cond   = (const float*)d_in[1];
  const float* W1     = (const float*)d_in[2];
  const float* atts1  = (const float*)d_in[3];
  const float* attd1  = (const float*)d_in[4];
  const float* b1     = (const float*)d_in[5];
  const float* W2     = (const float*)d_in[6];
  const float* atts2  = (const float*)d_in[7];
  const float* attd2  = (const float*)d_in[8];
  const float* b2     = (const float*)d_in[9];
  const float* outW   = (const float*)d_in[10];
  const float* outB   = (const float*)d_in[11];
  const int*   ei     = (const int*)d_in[12];
  const int*   batch  = (const int*)d_in[13];

  const int N = in_sizes[13];
  const int E = in_sizes[12] / 2;
  const long long Et = (long long)E + N;

  // ---- workspace carve-up (256B aligned) ----
  char* ws = (char*)d_ws;
  size_t off = 0;
  auto carve = [&](size_t bytes) { char* p = ws + off; off += (bytes + 255) & ~(size_t)255; return p; };
  unsigned short* A1   = (unsigned short*)carve((size_t)N * (IN_CH + CONDC) * 2);
  unsigned short* W1t  = (unsigned short*)carve((size_t)F1 * (IN_CH + CONDC) * 2);
  unsigned short* W2t  = (unsigned short*)carve((size_t)HIDC * F1 * 2);
  float* H1pre         = (float*)carve((size_t)N * F1 * 4);
  float* aS1           = (float*)carve((size_t)N * HEADS * 4);
  float* aD1           = (float*)carve((size_t)N * HEADS * 4);
  unsigned* m1         = (unsigned*)carve((size_t)N * HEADS * 4);
  float* s1            = (float*)carve((size_t)N * HEADS * 4);
  float* p1            = (float*)carve((size_t)Et * HEADS * 4);
  float* agg1          = (float*)carve((size_t)N * F1 * 4);
  unsigned short* H1bf = (unsigned short*)carve((size_t)N * F1 * 2);
  float* H2pre         = (float*)carve((size_t)N * HIDC * 4);
  float* aS2           = (float*)carve((size_t)N * 4);
  float* aD2           = (float*)carve((size_t)N * 4);
  unsigned* m2         = (unsigned*)carve((size_t)N * 4);
  float* s2            = (float*)carve((size_t)N * 4);
  float* p2            = (float*)carve((size_t)Et * 4);
  float* agg2          = (float*)carve((size_t)N * HIDC * 4);
  (void)ws_size;

  const int B = 256;
  auto blocks = [&](long long n) { return (unsigned)((n + B - 1) / B); };
  const unsigned ENC_NEG_INF = 0x007FFFFFu;         // fenc(-inf) = ~0xFF800000

  // init accumulators (deterministic every call)
  hipLaunchKernelGGL(k_fill_u32, dim3(blocks((long long)N * HEADS)), dim3(B), 0, stream, m1, ENC_NEG_INF, (long long)N * HEADS);
  hipLaunchKernelGGL(k_fill_u32, dim3(blocks((long long)N * HEADS)), dim3(B), 0, stream, (unsigned*)s1, 0u, (long long)N * HEADS);
  hipLaunchKernelGGL(k_fill_u32, dim3(blocks((long long)N * F1)), dim3(B), 0, stream, (unsigned*)agg1, 0u, (long long)N * F1);
  hipLaunchKernelGGL(k_fill_u32, dim3(blocks((long long)N)), dim3(B), 0, stream, m2, ENC_NEG_INF, (long long)N);
  hipLaunchKernelGGL(k_fill_u32, dim3(blocks((long long)N)), dim3(B), 0, stream, (unsigned*)s2, 0u, (long long)N);
  hipLaunchKernelGGL(k_fill_u32, dim3(blocks((long long)N * HIDC)), dim3(B), 0, stream, (unsigned*)agg2, 0u, (long long)N * HIDC);

  // weights -> bf16 transposed
  hipLaunchKernelGGL(k_convert_transpose, dim3(blocks((long long)(IN_CH + CONDC) * F1)), dim3(B), 0, stream, W1, W1t, IN_CH + CONDC, F1);
  hipLaunchKernelGGL(k_convert_transpose, dim3(blocks((long long)F1 * HIDC)), dim3(B), 0, stream, W2, W2t, F1, HIDC);

  // features
  hipLaunchKernelGGL(k_build_A1, dim3(blocks((long long)N * (IN_CH + CONDC))), dim3(B), 0, stream, x, cond, batch, A1, N);

  // GEMM1 (WMMA): [N x 1536] * [1536 x 512]
  {
    dim3 g((unsigned)((N + 127) / 128), F1 / 64);
    hipLaunchKernelGGL(k_gemm_wmma, g, dim3(256), 0, stream, A1, W1t, H1pre, N, IN_CH + CONDC, F1);
  }

  // layer-1 attention + aggregate
  hipLaunchKernelGGL(k_node_logits, dim3(blocks((long long)N * HEADS)), dim3(B), 0, stream, H1pre, atts1, attd1, aS1, aD1, N, HEADS, HIDC);
  hipLaunchKernelGGL(k_edge_max,    dim3(blocks(Et * HEADS)), dim3(B), 0, stream, ei, E, N, aS1, aD1, m1, HEADS);
  hipLaunchKernelGGL(k_edge_expsum, dim3(blocks(Et * HEADS)), dim3(B), 0, stream, ei, E, N, aS1, aD1, m1, p1, s1, HEADS);
  hipLaunchKernelGGL(k_edge_agg,    dim3(blocks(Et * HEADS * (HIDC / 4))), dim3(B), 0, stream, ei, E, N, H1pre, p1, s1, agg1, HEADS, HIDC);
  hipLaunchKernelGGL(k_bias_relu_bf16, dim3(blocks((long long)N * F1)), dim3(B), 0, stream, agg1, b1, H1bf, N, F1);

  // GEMM2 (WMMA): [N x 512] * [512 x 128]
  {
    dim3 g((unsigned)((N + 127) / 128), HIDC / 64);
    hipLaunchKernelGGL(k_gemm_wmma, g, dim3(256), 0, stream, H1bf, W2t, H2pre, N, F1, HIDC);
  }

  // layer-2 attention (H=1) + aggregate
  hipLaunchKernelGGL(k_node_logits, dim3(blocks((long long)N)), dim3(B), 0, stream, H2pre, atts2, attd2, aS2, aD2, N, 1, HIDC);
  hipLaunchKernelGGL(k_edge_max,    dim3(blocks(Et)), dim3(B), 0, stream, ei, E, N, aS2, aD2, m2, 1);
  hipLaunchKernelGGL(k_edge_expsum, dim3(blocks(Et)), dim3(B), 0, stream, ei, E, N, aS2, aD2, m2, p2, s2, 1);
  hipLaunchKernelGGL(k_edge_agg,    dim3(blocks(Et * (HIDC / 4))), dim3(B), 0, stream, ei, E, N, H2pre, p2, s2, agg2, 1, HIDC);

  // head
  hipLaunchKernelGGL(k_final, dim3(blocks((long long)N)), dim3(B), 0, stream, agg2, b2, outW, outB, (float*)d_out, N);
}
